// PairInteractionEmbeddingBlock_11605001634319
// MI455X (gfx1250) — compile-verified, matrix operand order checked by reading, not run on hardware
//
#include <hip/hip_runtime.h>
#include <math.h>

// MI455X / gfx1250 fused MACE-style pair-interaction block.
// wave32, WMMA f32<=f16 16x16x32 for all GEMM work; LDS-staged operands.
// Weights are pre-swizzled into per-lane WMMA fragment order in d_ws by a
// setup kernel so the hot kernel does zero f32->f16 conversion of weights.

typedef __attribute__((ext_vector_type(16))) _Float16 v16h;
typedef __attribute__((ext_vector_type(8)))  float    v8f;

#define TEDGES 16

// ---- LDS layout (bytes) ----
#define BF_OFF   0        // 9 slots * 64 kpairs * 16 t * 2 halfs = 36864 B (aliased by tpw later)
#define XS_OFF   36864    // 9 slots * 16 t * 66 halfs = 19008 B (row-padded)
#define H0_OFF   55872    // 16 t * 66 halfs = 2112 B
#define H1_OFF   57984    // 2112 B
#define YB_OFF   60096    // 16 t * 9 f32 = 576 B
#define SMEM_BYTES 60672

// ---- workspace fragment layout (in halfs): frag = 32 lanes * 16 halfs = 512 ----
// mixing A frags:   idx = (l*4+mt)*4+ks            -> [0, 48)
// mlp hidden B:     idx = 48 + layer*8 + nt*2 + ks -> [48, 72)
// mlp final B:      idx = 72 + nt*2 + ks           -> [72, 160)
#define NFRAG 160
#define REQ_WS ((size_t)NFRAG * 512 * 2)

// real CG constants (unit Frobenius norm, path norms folded at use sites)
#define CK2 0.31622776601683794f  // 1/sqrt(10)
#define CA2 0.18257418583505536f  // 1/sqrt(30)
#define CB2 0.36514837167011072f  // 2/sqrt(30)
#define CG1 0.23904572186687872f  // sqrt(2/35)
#define CG2 0.11952286093343936f  // CG1/2
#define CG3 0.20701966780270626f  // sqrt(3/70)
#define IS3 0.57735026918962576f  // 1/sqrt(3)
#define IS5 0.44721359549995794f  // 1/sqrt(5)

// K offset of vgpr-pair i (0..7) within a 32-wide K step of a 16-bit wmma fragment
__device__ __forceinline__ int frag_kbase(int i, int hi) {
  return ((i < 4) ? 0 : 16) + hi * 8 + (i & 3) * 2;
}

// ---------------- setup: pre-swizzle all weights into WMMA fragment order ----------------
__global__ __launch_bounds__(256)
void swizzle_weights_kernel(const float* __restrict__ W_src, const float* __restrict__ W_dst,
                            const float* __restrict__ w0, const float* __restrict__ w1,
                            const float* __restrict__ w2, const float* __restrict__ w3,
                            _Float16* __restrict__ ws)
{
  int gid = blockIdx.x * 256 + threadIdx.x;       // one (frag, lane) slot per thread
  if (gid >= NFRAG * 32) return;
  int fragIdx = gid >> 5;
  int lane = gid & 31;
  int nlo = lane & 15, hi = lane >> 4;
  _Float16* dst = ws + (size_t)fragIdx * 512 + lane * 16;
  if (fragIdx < 48) {                              // mixing A: [W_src | W_dst] * 0.125
    int ks = fragIdx & 3;
    int lm = fragIdx >> 2;
    int l = lm >> 2, mt = lm & 3;
    int u = mt * 16 + nlo;
    const float* wsrow = W_src + l * 4096 + u * 64;
    const float* wdrow = W_dst + l * 4096 + u * 64;
    #pragma unroll
    for (int i = 0; i < 8; ++i) {
      int kg = ks * 32 + frag_kbase(i, hi);
      const float* s = (kg < 64) ? (wsrow + kg) : (wdrow + (kg - 64));
      dst[2 * i]     = (_Float16)(s[0] * 0.125f);
      dst[2 * i + 1] = (_Float16)(s[1] * 0.125f);
    }
  } else if (fragIdx < 72) {                       // hidden-layer B frags
    int r = fragIdx - 48;
    int layer = r >> 3, q = r & 7, nt = q >> 1, ks = q & 1;
    const float* W = (layer == 0) ? w0 : ((layer == 1) ? w1 : w2);
    int n = nt * 16 + nlo;
    #pragma unroll
    for (int i = 0; i < 8; ++i) {
      int kg = ks * 32 + frag_kbase(i, hi);
      dst[2 * i]     = (_Float16)(W[kg * 64 + n] * 0.125f);
      dst[2 * i + 1] = (_Float16)(W[(kg + 1) * 64 + n] * 0.125f);
    }
  } else {                                         // final-layer (64x704) B frags
    int r = fragIdx - 72;
    int nt = r >> 1, ks = r & 1;
    int n = nt * 16 + nlo;
    #pragma unroll
    for (int i = 0; i < 8; ++i) {
      int kg = ks * 32 + frag_kbase(i, hi);
      dst[2 * i]     = (_Float16)(w3[kg * 704 + n] * 0.125f);
      dst[2 * i + 1] = (_Float16)(w3[(kg + 1) * 704 + n] * 0.125f);
    }
  }
}

// ---------------- fused main kernel ----------------
template <bool PRE>
__global__ __launch_bounds__(256)
void pair_block_kernel(const float* __restrict__ node_feats,
                       const int*   __restrict__ edge_index,
                       const float* __restrict__ edge_embed,
                       const float* __restrict__ edge_attrs,
                       const float* __restrict__ W_src,
                       const float* __restrict__ W_dst,
                       const float* __restrict__ mlp_w0,
                       const float* __restrict__ mlp_w1,
                       const float* __restrict__ mlp_w2,
                       const float* __restrict__ mlp_w3,
                       const _Float16* __restrict__ wfrag,
                       float* __restrict__ out,
                       int E)
{
  __shared__ __align__(16) unsigned char smem[SMEM_BYTES];
  _Float16* bf  = (_Float16*)(smem + BF_OFF);   // gathered feats, B-operand layout
  _Float16* xs  = (_Float16*)(smem + XS_OFF);   // mixed features x[slot][t][u] (row 66)
  _Float16* h0  = (_Float16*)(smem + H0_OFF);
  _Float16* h1  = (_Float16*)(smem + H1_OFF);
  float*    yb  = (float*)   (smem + YB_OFF);
  _Float16* tpw = (_Float16*)(smem + BF_OFF);   // tp weights [16][712], aliases bf after Phase B

  const int tid  = threadIdx.x;
  const int wave = tid >> 5;
  const int lane = tid & 31;
  const int nlo  = lane & 15;
  const int hi   = lane >> 4;
  const int edgeBase = blockIdx.x * TEDGES;

  // ---------------- Phase A: gather + stage operands ----------------
  {
    // 8 threads per (edge, side) node row; 128B per octet per iter (L2-resident table)
    int nodeSel = tid >> 3;       // 0..31
    int j       = tid & 7;
    int t       = nodeSel >> 1;
    int side    = nodeSel & 1;
    int e       = min(edgeBase + t, E - 1);
    int node    = edge_index[side * E + e];
    const float4* row4 = (const float4*)(node_feats + (size_t)node * 576);
    #pragma unroll 2
    for (int c = 0; c < 18; ++c) {
      float4 v = row4[c * 8 + j];
      int f0 = c * 32 + j * 4;
      float vals[4] = { v.x, v.y, v.z, v.w };
      #pragma unroll
      for (int q = 0; q < 4; ++q) {
        int f = f0 + q;
        float val = vals[q];
        int slot, vv;
        if (f < 64)        { slot = 0; vv = f; }
        else if (f < 256)  { int r = f - 64;  vv = r / 3; slot = 1 + (r - vv * 3); }
        else               { int r = f - 256; vv = r / 5; slot = 4 + (r - vv * 5); }
        int k = side * 64 + vv;                     // K = [src(64) | dst(64)]
        // pair-interleaved B layout: [slot][k>>1][t][k&1]
        bf[slot * 2048 + (k >> 1) * 32 + t * 2 + (k & 1)] = (_Float16)val;
      }
    }
  }
  {
    // h0 = edge_embed tile (f16, padded rows)
    int t  = tid >> 4;
    int k0 = (tid & 15) * 4;
    int e  = min(edgeBase + t, E - 1);
    const float4 v = *(const float4*)(edge_embed + (size_t)e * 64 + k0);
    h0[t * 66 + k0 + 0] = (_Float16)v.x;
    h0[t * 66 + k0 + 1] = (_Float16)v.y;
    h0[t * 66 + k0 + 2] = (_Float16)v.z;
    h0[t * 66 + k0 + 3] = (_Float16)v.w;
  }
  if (tid < 144) {
    int t = tid / 9, q = tid - t * 9;
    int e = min(edgeBase + t, E - 1);
    yb[t * 9 + q] = edge_attrs[(size_t)e * 9 + q];
  }
  __syncthreads();

  // ---------------- Phase B: equivariant mixing GEMM (M=64, K=128) ----------------
  // 36 output tiles = 4 Mtiles x (1+3+5) Ntiles, round-robin over 8 waves
  for (int tile = wave; tile < 36; tile += 8) {
    int mt, slot, l;
    if (tile < 4)       { mt = tile; slot = 0; l = 0; }
    else if (tile < 16) { int r = tile - 4;  mt = r / 3; slot = 1 + (r - mt * 3); l = 1; }
    else                { int r = tile - 16; mt = r / 5; slot = 4 + (r - mt * 5); l = 2; }
    int u = mt * 16 + nlo;
    const float* wsrow = W_src + l * 4096 + u * 64;
    const float* wdrow = W_dst + l * 4096 + u * 64;
    v8f acc = {};
    #pragma unroll
    for (int ks = 0; ks < 4; ++ks) {
      v16h a, b;
      if constexpr (PRE) {
        a = *(const v16h*)(wfrag + ((size_t)(((l * 4 + mt) * 4) + ks) * 512) + lane * 16);
      } else {
        #pragma unroll
        for (int i = 0; i < 8; ++i) {
          int kg = ks * 32 + frag_kbase(i, hi);
          const float* srcp = (kg < 64) ? (wsrow + kg) : (wdrow + (kg - 64));
          float2 f2 = *(const float2*)srcp;
          a[2 * i]     = (_Float16)(f2.x * 0.125f);
          a[2 * i + 1] = (_Float16)(f2.y * 0.125f);
        }
      }
      #pragma unroll
      for (int i = 0; i < 8; ++i) {
        int kg = ks * 32 + frag_kbase(i, hi);
        const _Float16* p = bf + slot * 2048 + (kg >> 1) * 32 + nlo * 2;
        b[2 * i]     = p[0];
        b[2 * i + 1] = p[1];
      }
      acc = __builtin_amdgcn_wmma_f32_16x16x32_f16(false, a, false, b, (short)0, acc, false, false);
    }
    #pragma unroll
    for (int r = 0; r < 8; ++r) {
      int uu = mt * 16 + r + hi * 8;               // D layout: M=r (+8 for hi half)
      xs[slot * 1056 + nlo * 66 + uu] = (_Float16)acc[r];
    }
  }
  __syncthreads();

  // ---------------- Phase C: radial MLP (M=16 edges) ----------------
  _Float16* hcur  = h0;
  _Float16* hnext = h1;
  #pragma unroll 1
  for (int layer = 0; layer < 3; ++layer) {
    const float* W = (layer == 0) ? mlp_w0 : ((layer == 1) ? mlp_w1 : mlp_w2);
    if (wave < 4) {
      int n = wave * 16 + nlo;
      v8f acc = {};
      #pragma unroll
      for (int ks = 0; ks < 2; ++ks) {
        v16h a, b;
        #pragma unroll
        for (int i = 0; i < 8; ++i) {
          int kg = ks * 32 + frag_kbase(i, hi);
          const _Float16* pa = hcur + nlo * 66 + kg;     // A: M = edge t
          a[2 * i]     = pa[0];
          a[2 * i + 1] = pa[1];
        }
        if constexpr (PRE) {
          b = *(const v16h*)(wfrag + (size_t)(48 + layer * 8 + wave * 2 + ks) * 512 + lane * 16);
        } else {
          #pragma unroll
          for (int i = 0; i < 8; ++i) {
            int kg = ks * 32 + frag_kbase(i, hi);
            b[2 * i]     = (_Float16)(W[kg * 64 + n] * 0.125f);
            b[2 * i + 1] = (_Float16)(W[(kg + 1) * 64 + n] * 0.125f);
          }
        }
        acc = __builtin_amdgcn_wmma_f32_16x16x32_f16(false, a, false, b, (short)0, acc, false, false);
      }
      #pragma unroll
      for (int r = 0; r < 8; ++r) {
        int te = r + hi * 8;
        float v = acc[r];
        v = v / (1.0f + __expf(-v));                     // silu
        hnext[te * 66 + n] = (_Float16)v;
      }
    }
    __syncthreads();
    _Float16* tmp = hcur; hcur = hnext; hnext = tmp;
  }
  // final layer: [16,64] @ [64,704] -> tpw (aliases bf; safe after Phase B barrier)
  for (int nt = wave; nt < 44; nt += 8) {
    int n = nt * 16 + nlo;
    v8f acc = {};
    #pragma unroll
    for (int ks = 0; ks < 2; ++ks) {
      v16h a, b;
      #pragma unroll
      for (int i = 0; i < 8; ++i) {
        int kg = ks * 32 + frag_kbase(i, hi);
        const _Float16* pa = hcur + nlo * 66 + kg;
        a[2 * i]     = pa[0];
        a[2 * i + 1] = pa[1];
      }
      if constexpr (PRE) {
        b = *(const v16h*)(wfrag + (size_t)(72 + nt * 2 + ks) * 512 + lane * 16);
      } else {
        #pragma unroll
        for (int i = 0; i < 8; ++i) {
          int kg = ks * 32 + frag_kbase(i, hi);
          b[2 * i]     = (_Float16)(mlp_w3[kg * 704 + n] * 0.125f);
          b[2 * i + 1] = (_Float16)(mlp_w3[(kg + 1) * 704 + n] * 0.125f);
        }
      }
      acc = __builtin_amdgcn_wmma_f32_16x16x32_f16(false, a, false, b, (short)0, acc, false, false);
    }
    #pragma unroll
    for (int r = 0; r < 8; ++r) {
      int te = r + hi * 8;
      tpw[te * 712 + n] = (_Float16)acc[r];
    }
  }
  __syncthreads();

  // ---------------- Phase D: CG tensor product + weighted scatter to output ----------------
  // wave owns edges 2w, 2w+1; lane sweeps u -> contiguous per-lane output runs (coalesced)
  #pragma unroll
  for (int ii = 0; ii < 2; ++ii) {
    int t = wave * 2 + ii;
    int e = edgeBase + t;
    float y0 = yb[t * 9 + 0];
    float y1[3], y2[5];
    #pragma unroll
    for (int m = 0; m < 3; ++m) y1[m] = yb[t * 9 + 1 + m];
    #pragma unroll
    for (int m = 0; m < 5; ++m) y2[m] = yb[t * 9 + 4 + m];
    #pragma unroll
    for (int jj = 0; jj < 2; ++jj) {
      int u = lane + 32 * jj;
      float x0 = (float)xs[t * 66 + u];
      float x1[3], x2[5];
      #pragma unroll
      for (int m = 0; m < 3; ++m) x1[m] = (float)xs[(1 + m) * 1056 + t * 66 + u];
      #pragma unroll
      for (int m = 0; m < 5; ++m) x2[m] = (float)xs[(4 + m) * 1056 + t * 66 + u];
      float w[11];
      #pragma unroll
      for (int p = 0; p < 11; ++p) w[p] = (float)tpw[t * 712 + p * 64 + u];

      // l_out = 0  (paths 000, 110, 220), norm 1/sqrt(3)
      float d11 = x1[0] * y1[0] + x1[1] * y1[1] + x1[2] * y1[2];
      float d22 = x2[0] * y2[0] + x2[1] * y2[1] + x2[2] * y2[2] + x2[3] * y2[3] + x2[4] * y2[4];
      float o0 = IS3 * (w[0] * x0 * y0 + w[1] * IS3 * d11 + w[2] * IS5 * d22);

      // l_out = 1  (paths 011, 101, 121, 211), norm 1/2
      float t121_0 = CK2 * (y2[0] * x1[2] + y2[1] * x1[1]) - CA2 * y2[2] * x1[0] - CK2 * y2[4] * x1[0];
      float t121_1 = CK2 * (y2[1] * x1[0] + y2[3] * x1[2]) + CB2 * y2[2] * x1[1];
      float t121_2 = CK2 * (y2[0] * x1[0] + y2[3] * x1[1] + y2[4] * x1[2]) - CA2 * y2[2] * x1[2];
      float t211_0 = CK2 * (x2[0] * y1[2] + x2[1] * y1[1]) - CA2 * x2[2] * y1[0] - CK2 * x2[4] * y1[0];
      float t211_1 = CK2 * (x2[1] * y1[0] + x2[3] * y1[2]) + CB2 * x2[2] * y1[1];
      float t211_2 = CK2 * (x2[0] * y1[0] + x2[3] * y1[1] + x2[4] * y1[2]) - CA2 * x2[2] * y1[2];
      float o1[3];
      o1[0] = 0.5f * (IS3 * (w[3] * x0 * y1[0] + w[4] * y0 * x1[0]) + w[5] * t121_0 + w[6] * t211_0);
      o1[1] = 0.5f * (IS3 * (w[3] * x0 * y1[1] + w[4] * y0 * x1[1]) + w[5] * t121_1 + w[6] * t211_1);
      o1[2] = 0.5f * (IS3 * (w[3] * x0 * y1[2] + w[4] * y0 * x1[2]) + w[5] * t121_2 + w[6] * t211_2);

      // l_out = 2  (paths 022, 112, 202, 222), norm 1/2
      float t112_0 = CK2 * (x1[0] * y1[2] + x1[2] * y1[0]);
      float t112_1 = CK2 * (x1[0] * y1[1] + x1[1] * y1[0]);
      float t112_2 = -CA2 * x1[0] * y1[0] + CB2 * x1[1] * y1[1] - CA2 * x1[2] * y1[2];
      float t112_3 = CK2 * (x1[1] * y1[2] + x1[2] * y1[1]);
      float t112_4 = CK2 * (x1[2] * y1[2] - x1[0] * y1[0]);
      float t222_0 = -CG1 * (x2[0] * y2[2] + x2[2] * y2[0]) + CG3 * (x2[1] * y2[3] + x2[3] * y2[1]);
      float t222_1 =  CG2 * (x2[1] * y2[2] + x2[2] * y2[1]) - CG3 * (x2[1] * y2[4] + x2[4] * y2[1])
                    + CG3 * (x2[0] * y2[3] + x2[3] * y2[0]);
      float t222_2 =  CG1 * (x2[2] * y2[2] - x2[0] * y2[0] - x2[4] * y2[4])
                    + CG2 * (x2[1] * y2[1] + x2[3] * y2[3]);
      float t222_3 =  CG3 * (x2[0] * y2[1] + x2[1] * y2[0]) + CG2 * (x2[3] * y2[2] + x2[2] * y2[3])
                    + CG3 * (x2[3] * y2[4] + x2[4] * y2[3]);
      float t222_4 = -CG1 * (x2[4] * y2[2] + x2[2] * y2[4]) - CG3 * x2[1] * y2[1] + CG3 * x2[3] * y2[3];
      float o2[5];
      o2[0] = 0.5f * (IS5 * (w[7] * x0 * y2[0] + w[9] * y0 * x2[0]) + w[8] * t112_0 + w[10] * t222_0);
      o2[1] = 0.5f * (IS5 * (w[7] * x0 * y2[1] + w[9] * y0 * x2[1]) + w[8] * t112_1 + w[10] * t222_1);
      o2[2] = 0.5f * (IS5 * (w[7] * x0 * y2[2] + w[9] * y0 * x2[2]) + w[8] * t112_2 + w[10] * t222_2);
      o2[3] = 0.5f * (IS5 * (w[7] * x0 * y2[3] + w[9] * y0 * x2[3]) + w[8] * t112_3 + w[10] * t222_3);
      o2[4] = 0.5f * (IS5 * (w[7] * x0 * y2[4] + w[9] * y0 * x2[4]) + w[8] * t112_4 + w[10] * t222_4);

      if (e < E) {
        float* ob = out + (size_t)e * 576;
        ob[u] = o0;
        ob[64 + u * 3 + 0] = o1[0];
        ob[64 + u * 3 + 1] = o1[1];
        ob[64 + u * 3 + 2] = o1[2];
        ob[256 + u * 5 + 0] = o2[0];
        ob[256 + u * 5 + 1] = o2[1];
        ob[256 + u * 5 + 2] = o2[2];
        ob[256 + u * 5 + 3] = o2[3];
        ob[256 + u * 5 + 4] = o2[4];
      }
    }
  }
}

extern "C" void kernel_launch(void* const* d_in, const int* in_sizes, int n_in,
                              void* d_out, int out_size, void* d_ws, size_t ws_size,
                              hipStream_t stream) {
  const float* node_feats = (const float*)d_in[0];
  const int*   edge_index = (const int*)  d_in[1];
  const float* edge_embed = (const float*)d_in[2];
  const float* edge_attrs = (const float*)d_in[3];
  const float* W_src      = (const float*)d_in[4];
  const float* W_dst      = (const float*)d_in[5];
  const float* mlp_w0     = (const float*)d_in[6];
  const float* mlp_w1     = (const float*)d_in[7];
  const float* mlp_w2     = (const float*)d_in[8];
  const float* mlp_w3     = (const float*)d_in[9];
  float* out = (float*)d_out;
  (void)n_in; (void)out_size;

  int E = in_sizes[1] / 2;                 // edge_index is [2, E]
  dim3 grid((E + TEDGES - 1) / TEDGES), block(256);

  if (ws_size >= REQ_WS) {
    _Float16* wfrag = (_Float16*)d_ws;
    dim3 sgrid((NFRAG * 32 + 255) / 256);
    swizzle_weights_kernel<<<sgrid, block, 0, stream>>>(W_src, W_dst, mlp_w0, mlp_w1, mlp_w2, mlp_w3, wfrag);
    pair_block_kernel<true><<<grid, block, 0, stream>>>(node_feats, edge_index, edge_embed, edge_attrs,
                                                        W_src, W_dst, mlp_w0, mlp_w1, mlp_w2, mlp_w3,
                                                        (const _Float16*)wfrag, out, E);
  } else {
    pair_block_kernel<false><<<grid, block, 0, stream>>>(node_feats, edge_index, edge_embed, edge_attrs,
                                                         W_src, W_dst, mlp_w0, mlp_w1, mlp_w2, mlp_w3,
                                                         (const _Float16*)d_ws, out, E);
  }
}
